// NMMO3_66614942761593
// MI455X (gfx1250) — matrix operand order repaired and checked by reading, not compiled, not run on hardware
//
#include <hip/hip_runtime.h>

// ---------------------------------------------------------------------------
// NMMO3 policy forward for MI455X (gfx1250, wave32, WMMA).
//  obs(int) -> sparse conv1 (gather-sum) -> relu -> conv2 -> concat features
//  -> f16 WMMA proj GEMM + relu -> fused WMMA actor/value heads.
// ---------------------------------------------------------------------------

typedef _Float16 v16h __attribute__((ext_vector_type(16)));
typedef float    v8f  __attribute__((ext_vector_type(8)));

namespace {
constexpr int Hh = 11, Ww = 15, NF = 10, Pp = 47, NR = 10;
constexpr int OBSLEN = Hh * Ww * NF + Pp + NR;   // 1707
constexpr int MAPLEN = Hh * Ww * NF;             // 1650
constexpr int KRAW   = 1689;                     // PROJ_IN
constexpr int KP     = 1696;                     // padded to 53*32
constexpr int HID    = 256;
constexpr int NACT   = 26;
constexpr int TB     = 16;                       // batch elems per feat block

// workspace layout (bytes, 256-aligned)
constexpr size_t OFF_W1T  = 0;                      // 1475*64*4  = 377600
constexpr size_t OFF_W2T  = 377600;                 // 576*64*4   = 147456
constexpr size_t OFF_PW16 = 525056;                 // 256*1696*2 = 868352
constexpr size_t OFF_HW16 = 1393408;                // 32*256*2   = 16384
constexpr size_t OFF_X16  = 1409792;                // batch*1696*2
}

// ---------------------------------------------------------------------------
// Kernel 0: weight reshaping.  W1T/W2T transposed to [k][oc] so the 64 oc
// lanes make coalesced 256B reads; proj/head weights converted to f16 with
// K-contiguous rows (what each WMMA B-lane consumes as 32 contiguous bytes).
// ---------------------------------------------------------------------------
__global__ __launch_bounds__(256) void prep_kernel(
    const float* __restrict__ c1w, const float* __restrict__ c2w,
    const float* __restrict__ pw,  const float* __restrict__ aw,
    const float* __restrict__ vw,
    float* __restrict__ W1T, float* __restrict__ W2T,
    _Float16* __restrict__ PW16, _Float16* __restrict__ HW16)
{
    constexpr int NW1 = 1475 * 64, NW2 = 576 * 64, NPW = HID * KP, NHW = 32 * HID;
    constexpr int TOTAL = NW1 + NW2 + NPW + NHW;
    for (int i = blockIdx.x * 256 + threadIdx.x; i < TOTAL; i += gridDim.x * 256) {
        if (i < NW1) {
            int t = i >> 6, oc = i & 63;
            W1T[i] = c1w[oc * 1475 + t];
        } else if (i < NW1 + NW2) {
            int i2 = i - NW1, t = i2 >> 6, oc = i2 & 63;
            W2T[i2] = c2w[oc * 576 + t];
        } else if (i < NW1 + NW2 + NPW) {
            int i3 = i - NW1 - NW2, n = i3 / KP, k = i3 - n * KP;
            PW16[i3] = (k < KRAW) ? (_Float16)pw[n * KRAW + k] : (_Float16)0.f;
        } else {
            int i4 = i - NW1 - NW2 - NPW, n = i4 >> 8, k = i4 & 255;
            float v = (n < NACT) ? aw[n * HID + k] : (n == NACT ? vw[k] : 0.f);
            HW16[i4] = (_Float16)v;
        }
    }
}

// ---------------------------------------------------------------------------
// Kernel 1: feature builder. One block = 16 batch elements.
// conv1 exploits one-hot structure: out = bias + sum of 250 selected weight
// rows (coefficient exactly 1.0) -> no dense 59-ch map ever materialized.
// ---------------------------------------------------------------------------
__global__ __launch_bounds__(256) void feat_kernel(
    const int*   __restrict__ obs,
    const float* __restrict__ conv1_b,
    const float* __restrict__ conv2_b,
    const float* __restrict__ embed,
    const float* __restrict__ W1T,
    const float* __restrict__ W2T,
    _Float16*    __restrict__ X)
{
    __shared__ unsigned char mapB[TB * MAPLEN];   // 26400 B
    __shared__ unsigned char plB [TB * Pp];       //   752 B
    __shared__ unsigned char rwB [TB * NR];       //   160 B
    __shared__ _Float16      r1  [TB * 12 * 64];  // 24576 B  (relu(conv1))

    const int tid = threadIdx.x;
    const int b0  = blockIdx.x * TB;

    // ---- stage observations as bytes ----
    for (int i = tid; i < TB * OBSLEN; i += 256) {
        int e = i / OBSLEN, j = i - e * OBSLEN;
        unsigned char c = (unsigned char)obs[(size_t)(b0 + e) * OBSLEN + j];
        if (j < MAPLEN)            mapB[e * MAPLEN + j] = c;
        else if (j < MAPLEN + Pp)  plB[e * Pp + (j - MAPLEN)] = c;
        else                       rwB[e * NR + (j - MAPLEN - Pp)] = c;
    }
    __syncthreads();

    const int g = tid >> 6;     // 4 groups of 64 lanes (lane = output channel)
    const int oc = tid & 63;

    // ---- conv1: sparse multihot gather-sum, stride 3, out (64,3,4) ----
    static constexpr int OFFS[10] = {0, 4, 8, 25, 30, 33, 38, 43, 48, 55};
    for (int item = g; item < TB * 12; item += 4) {
        int e = item / 12, p = item - e * 12;
        int oh = p >> 2, ow = p & 3;
        // two interleaved partial sums -> more loads in flight per clause
        float acc0 = conv1_b[oc];
        float acc1 = 0.f;
        const unsigned char* m = &mapB[e * MAPLEN];
#pragma unroll
        for (int f = 0; f < 10; ++f) {
            const int base = OFFS[f];
            for (int kh = 0; kh < 5; ++kh) {
                int h = oh * 3 + kh;
#pragma unroll
                for (int kw = 0; kw < 5; ++kw) {
                    int w = ow * 3 + kw;
                    int v = m[(h * Ww + w) * NF + f];      // LDS broadcast
                    float wt = W1T[(size_t)((base + v) * 25 + kh * 5 + kw) * 64 + oc];
                    if (kw & 1) acc1 += wt; else acc0 += wt;
                }
            }
        }
        float acc = acc0 + acc1;
        r1[(e * 12 + p) * 64 + oc] = (_Float16)(acc > 0.f ? acc : 0.f);
    }
    __syncthreads();

    // ---- conv2 3x3 -> map_feat (X[0..127], layout oc*2 + p2) ----
    for (int item = g; item < TB * 2; item += 4) {
        int e = item >> 1, p2 = item & 1;
        float acc = conv2_b[oc];
        for (int ic = 0; ic < 64; ++ic) {
#pragma unroll
            for (int kh = 0; kh < 3; ++kh) {
#pragma unroll
                for (int kw = 0; kw < 3; ++kw) {
                    float r = (float)r1[(e * 12 + kh * 4 + p2 + kw) * 64 + ic];
                    acc += r * W2T[(size_t)((ic * 3 + kh) * 3 + kw) * 64 + oc];
                }
            }
        }
        X[(size_t)(b0 + e) * KP + oc * 2 + p2] = (_Float16)acc;
    }

    // ---- embeddings: X[128 + j*32 + u] = embed[val][u] ----
    for (int i = tid; i < TB * Pp * 32; i += 256) {
        int e = i / (Pp * 32), r = i - e * (Pp * 32);
        int j = r >> 5, u = r & 31;
        int v = plB[e * Pp + j];
        X[(size_t)(b0 + e) * KP + 128 + j * 32 + u] = (_Float16)embed[v * 32 + u];
    }

    // ---- raw player (47) + reward (10) + zero pad (7): X[1632..1695] ----
    for (int i = tid; i < TB * 64; i += 256) {
        int e = i >> 6, j = i & 63;
        float val = 0.f;
        if (j < Pp)           val = (float)plB[e * Pp + j];
        else if (j < Pp + NR) val = (float)rwB[e * NR + (j - Pp)];
        X[(size_t)(b0 + e) * KP + 1632 + j] = (_Float16)val;
    }
}

// ---------------------------------------------------------------------------
// Kernel 2: fused proj GEMM (relu) + actor/value heads via WMMA f16.
// Block = 64 batch rows, 8 waves. Wave (wM=w>>1, wN=w&1) owns 8 accum tiles
// covering a 16x128 slice of the 64x256 HDN tile. HDN lives only in LDS.
// A-layout: lane l holds row M=l%16, K = kb..kb+15 contiguous,
// kb = (l<16 ? 0 : 16). B-layout: lane l holds col N=l%16, K = kk..kk+15
// contiguous, kk = (l<16 ? 0 : 16).
// ---------------------------------------------------------------------------
__global__ __launch_bounds__(256) void gemm_kernel(
    const _Float16* __restrict__ X,
    const _Float16* __restrict__ PW,     // (256, 1696) f16, n-major
    const float*    __restrict__ proj_b,
    const _Float16* __restrict__ HW,     // (32, 256) f16: actor rows, value, pad
    const float*    __restrict__ actor_b,
    const float*    __restrict__ value_b,
    float*          __restrict__ out,
    int batch)
{
    __shared__ __align__(16) _Float16 hdn[64][264];   // padded stride vs banks

    const int tid  = threadIdx.x;
    const int wv   = tid >> 5;
    const int lane = tid & 31;
    const int lrow = lane & 15;
    const int lhi  = lane >> 4;
    const int wM   = wv >> 1;      // 0..3 -> 16-row stripe
    const int wN   = wv & 1;       // 0..1 -> 128-col stripe
    const long Mbase = (long)blockIdx.x * 64;

    // ---------------- proj: HDN = relu(X @ PW^T + b) ----------------
    v8f acc[8] = {};
    const int m = (int)Mbase + wM * 16 + lrow;
    const _Float16* xrow = X + (size_t)m * KP;

    for (int kt = 0; kt < KP; kt += 32) {
        // prefetch next K-chunk of activations and weights into cache
        if (kt + 32 < KP) {
            __builtin_prefetch(xrow + kt + 32 + lhi * 8, 0, 1);
            __builtin_prefetch(PW + (size_t)((wN * 8) * 16 + lrow) * KP + kt + 32 + lhi * 16, 0, 1);
        }
        v16h a;
        __builtin_memcpy(&a,              xrow + kt + lhi * 16,      16);
        __builtin_memcpy((char*)&a + 16,  xrow + kt + lhi * 16 + 8, 16);
#pragma unroll
        for (int t = 0; t < 8; ++t) {
            int n = (wN * 8 + t) * 16 + lrow;
            v16h b;
            __builtin_memcpy(&b, PW + (size_t)n * KP + kt + lhi * 16, 32);
            acc[t] = __builtin_amdgcn_wmma_f32_16x16x32_f16(
                false, a, false, b, (short)0, acc[t], false, false);
        }
    }

#pragma unroll
    for (int t = 0; t < 8; ++t) {
        int n = (wN * 8 + t) * 16 + lrow;
        float bias = proj_b[n];
#pragma unroll
        for (int r = 0; r < 8; ++r) {
            float v = acc[t][r] + bias;
            v = v > 0.f ? v : 0.f;                 // relu
            hdn[wM * 16 + r + (lhi ? 8 : 0)][n] = (_Float16)v;
        }
    }
    __syncthreads();

    // ---------------- heads: (64x256) @ (256x32) ----------------
    const int mt = wv >> 1, nt = wv & 1;          // 4x2 tiles, one per wave
    v8f c2 = {};
    for (int kt = 0; kt < HID; kt += 32) {
        v16h a;
        const _Float16* hrow = &hdn[mt * 16 + lrow][0];
        __builtin_memcpy(&a,             hrow + kt + lhi * 16,      16);
        __builtin_memcpy((char*)&a + 16, hrow + kt + lhi * 16 + 8, 16);
        int n = nt * 16 + lrow;
        v16h b;
        __builtin_memcpy(&b, HW + (size_t)n * HID + kt + lhi * 16, 32);
        c2 = __builtin_amdgcn_wmma_f32_16x16x32_f16(
            false, a, false, b, (short)0, c2, false, false);
    }

    const int n = nt * 16 + lrow;
    const float bias = (n < NACT) ? actor_b[n] : (n == NACT ? value_b[0] : 0.f);
#pragma unroll
    for (int r = 0; r < 8; ++r) {
        long brow = Mbase + mt * 16 + r + (lhi ? 8 : 0);
        float v = c2[r] + bias;
        if (n < NACT)        out[brow * NACT + n] = v;          // actions
        else if (n == NACT)  out[(long)batch * NACT + brow] = v; // value
    }
}

// ---------------------------------------------------------------------------
extern "C" void kernel_launch(void* const* d_in, const int* in_sizes, int n_in,
                              void* d_out, int out_size, void* d_ws, size_t ws_size,
                              hipStream_t stream) {
    const int*   obs     = (const int*)  d_in[0];
    const float* conv1_w = (const float*)d_in[1];
    const float* conv1_b = (const float*)d_in[2];
    const float* conv2_w = (const float*)d_in[3];
    const float* conv2_b = (const float*)d_in[4];
    const float* embed   = (const float*)d_in[5];
    const float* proj_w  = (const float*)d_in[6];
    const float* proj_b  = (const float*)d_in[7];
    const float* actor_w = (const float*)d_in[8];
    const float* actor_b = (const float*)d_in[9];
    const float* value_w = (const float*)d_in[10];
    const float* value_b = (const float*)d_in[11];

    const int batch = in_sizes[0] / OBSLEN;   // 16384

    char* ws = (char*)d_ws;
    float*    W1T  = (float*)   (ws + OFF_W1T);
    float*    W2T  = (float*)   (ws + OFF_W2T);
    _Float16* PW16 = (_Float16*)(ws + OFF_PW16);
    _Float16* HW16 = (_Float16*)(ws + OFF_HW16);
    _Float16* X16  = (_Float16*)(ws + OFF_X16);

    prep_kernel<<<1024, 256, 0, stream>>>(conv1_w, conv2_w, proj_w, actor_w,
                                          value_w, W1T, W2T, PW16, HW16);
    feat_kernel<<<batch / TB, 256, 0, stream>>>(obs, conv1_b, conv2_b, embed,
                                                W1T, W2T, X16);
    gemm_kernel<<<batch / 64, 256, 0, stream>>>(X16, PW16, proj_b, HW16,
                                                actor_b, value_b,
                                                (float*)d_out, batch);
}
